// DIN_11562051961035
// MI455X (gfx1250) — matrix operand order repaired and checked by reading, not compiled, not run on hardware
//
#include <hip/hip_runtime.h>

typedef __attribute__((ext_vector_type(16))) _Float16 v16h;
typedef __attribute__((ext_vector_type(8)))  _Float16 v8h;
typedef __attribute__((ext_vector_type(8)))  float    v8f;

#define T_LEN      200
#define T_PAD      208          // 13 * 16
#define MT_TILES   13
#define D_EMB      64
#define K_IN       256          // 4*D (original attn_in width)
#define K_RED      128          // reduced K: [h, h*t]
#define H_A        128
#define LDS_STRIDE 136          // 128 + 8 halves: 272B row stride, 16B aligned, conflict-free

// ---------------------------------------------------------------------------
// Prep:
//   Wm    f16 [H_A][K_RED] : Wm[n][k<64]    = W_h[k][n] + W_d[k][n]
//                            Wm[n][64+k]    = W_p[k][n]       (transposed, fp16)
//   Wdiff f32 [64][H_A]    : Wdiff[d][n]    = W_t[d][n] - W_d[d][n]
// where W1a rows [0:64)=W_h, [64:128)=W_t, [128:192)=W_d, [192:256)=W_p.
// ---------------------------------------------------------------------------
__global__ void __launch_bounds__(256)
din_prep_weights(const float* __restrict__ W1a,
                 _Float16* __restrict__ Wm,
                 float* __restrict__ Wdiff) {
    int i = blockIdx.x * blockDim.x + threadIdx.x;
    if (i < H_A * K_RED) {
        int n = i >> 7;          // hidden unit 0..127
        int k = i & 127;         // reduced input feature
        float v;
        if (k < 64)
            v = W1a[k * H_A + n] + W1a[(128 + k) * H_A + n];   // W_h + W_d
        else
            v = W1a[(192 + (k - 64)) * H_A + n];                // W_p
        Wm[n * K_RED + k] = (_Float16)v;
    }
    if (i < 64 * H_A) {
        int d = i >> 7;
        int n = i & 127;
        Wdiff[d * H_A + n] = W1a[(64 + d) * H_A + n] - W1a[(128 + d) * H_A + n];
    }
}

// ---------------------------------------------------------------------------
// Main: one block (8 waves) per batch row
// ---------------------------------------------------------------------------
__global__ void __launch_bounds__(256)
din_main(const int*   __restrict__ hist_ids,
         const int*   __restrict__ hist_mask,
         const int*   __restrict__ tgt_ids,
         const float* __restrict__ emb,
         const float* __restrict__ b1a,
         const float* __restrict__ W2a,
         const float* __restrict__ b2a,
         const float* __restrict__ W1p,
         const float* __restrict__ b1p,
         const float* __restrict__ W2p,
         const float* __restrict__ b2p,
         const _Float16* __restrict__ Wm,
         const float*    __restrict__ Wdiff,
         float* __restrict__ out)
{
    __shared__ __align__(16) _Float16 sA[T_PAD][LDS_STRIDE];  // [h, h*t] in f16
    __shared__ float sT[D_EMB];
    __shared__ float sW2a[H_A];
    __shared__ float sBias[H_A];       // b1a + t @ Wdiff (per-batch folded bias)
    __shared__ float sScores[T_PAD];
    __shared__ float sRed[256];
    __shared__ float sU[4][D_EMB];
    __shared__ float sPin[K_IN];

    const int tid = threadIdx.x;
    const int b   = blockIdx.x;
    const float NEG_INF = -__builtin_inff();

    // ---- stage target embedding, small weights, zero scores ----
    if (tid < D_EMB) {
        int t_id = tgt_ids[b];
        sT[tid] = emb[(size_t)t_id * D_EMB + tid];
    }
    if (tid < H_A) sW2a[tid] = W2a[tid];
    if (tid < T_PAD) sScores[tid] = 0.0f;
    __syncthreads();

    // ---- per-batch folded bias: sBias[n] = b1a[n] + sum_d t[d]*Wdiff[d][n] ----
    if (tid < H_A) {
        float acc = b1a[tid];
        #pragma unroll 8
        for (int d = 0; d < D_EMB; ++d)
            acc += sT[d] * Wdiff[d * H_A + tid];
        sBias[tid] = acc;
    }

    // ---- build reduced A = [h, h*t] in LDS as f16 ----
    for (int idx = tid; idx < T_PAD * D_EMB; idx += 256) {
        int r = idx >> 6;
        int d = idx & 63;
        float hv = 0.0f;
        if (r < T_LEN) {
            int id = hist_ids[(size_t)b * T_LEN + r];
            hv = emb[(size_t)id * D_EMB + d];
        }
        sA[r][d]         = (_Float16)hv;
        sA[r][D_EMB + d] = (_Float16)(hv * sT[d]);
    }
    __syncthreads();

    // ---- WMMA GEMM: [T_PAD,128] x [128,128], fused bias+relu+W2a+score ----
    {
        const int lane  = tid & 31;
        const int wv    = tid >> 5;     // wave owns hidden cols [wv*16, wv*16+16)
        const int nloc  = lane & 15;
        const int lhalf = lane >> 4;    // 0 or 1

        // Preload B fragments (K x N, N = wv*16 + nloc per lane).
        // Lanes 0-15 hold K k0..k0+15, lanes 16-31 hold K k0+16..k0+31.
        v16h bfrag[4];
        const _Float16* wrow = Wm + (size_t)(wv * 16 + nloc) * K_RED;
        #pragma unroll
        for (int ks = 0; ks < 4; ++ks)
            bfrag[ks] = *reinterpret_cast<const v16h*>(wrow + ks * 32 + lhalf * 16);

        const float w2   = sW2a[wv * 16 + nloc];
        const float bias = sBias[wv * 16 + nloc];

        for (int mt = 0; mt < MT_TILES; ++mt) {
            v8f c = {};
            // A fragment: lane holds row M = mt*16 + nloc; contiguous K layout matching B:
            // lanes 0-15 K {0..15}+ks*32, lanes 16-31 K {16..31}+ks*32.
            const _Float16* arow  = &sA[mt * 16 + nloc][0];
            const int       abase = lhalf * 16;
            #pragma unroll
            for (int ks = 0; ks < 4; ++ks) {
                v8h lo = *reinterpret_cast<const v8h*>(arow + ks * 32 + abase);
                v8h hi = *reinterpret_cast<const v8h*>(arow + ks * 32 + 8 + abase);
                v16h a = __builtin_shufflevector(lo, hi,
                          0,1,2,3,4,5,6,7,8,9,10,11,12,13,14,15);
                c = __builtin_amdgcn_wmma_f32_16x16x32_f16(
                        false, a, false, bfrag[ks], (short)0, c, false, false);
            }
            // C layout: c[j] at lane L = hidden[mt*16 + j + lhalf*8][wv*16 + nloc]
            // Fused epilogue: relu(c + bias) * W2a[n], summed over n via LDS
            // atomics (16-way collisions resolved inside the ds_add unit).
            #pragma unroll
            for (int j = 0; j < 8; ++j) {
                float hpre = c[j] + bias;
                float v = (hpre > 0.0f ? hpre : 0.0f) * w2;
                atomicAdd(&sScores[mt * 16 + j + lhalf * 8], v);
            }
        }
    }
    __syncthreads();

    // ---- masked softmax over T ----
    {
        const int m = tid;
        bool  valid = false;
        float sv    = NEG_INF;
        if (m < T_LEN) {
            valid = hist_mask[(size_t)b * T_LEN + m] != 0;
            if (valid) sv = sScores[m] + b2a[0];
        }
        sRed[tid] = sv;
        __syncthreads();
        for (int off = 128; off > 0; off >>= 1) {
            if (tid < off) sRed[tid] = fmaxf(sRed[tid], sRed[tid + off]);
            __syncthreads();
        }
        float mx = sRed[0];
        __syncthreads();

        float e = 0.0f;
        if (valid) e = __expf(sv - mx);
        sRed[tid] = e;
        __syncthreads();
        for (int off = 128; off > 0; off >>= 1) {
            if (tid < off) sRed[tid] += sRed[tid + off];
            __syncthreads();
        }
        float Z = fmaxf(sRed[0], 1e-8f);
        __syncthreads();
        if (m < T_PAD) sScores[m] = (m < T_LEN) ? e / Z : 0.0f;
    }
    __syncthreads();

    // ---- u = sum_t w_t * h_t (re-gather, L2-resident) ----
    {
        const int d = tid & 63;
        const int g = tid >> 6;     // 4 row-groups
        float acc = 0.0f;
        for (int r = g; r < T_LEN; r += 4) {
            float wgt = sScores[r];
            if (wgt != 0.0f) {
                int id = hist_ids[(size_t)b * T_LEN + r];
                acc += wgt * emb[(size_t)id * D_EMB + d];
            }
        }
        sU[g][d] = acc;
    }
    __syncthreads();

    // ---- pred_in = [u, t, u-t, u*t] ----
    if (tid < D_EMB) {
        float u  = sU[0][tid] + sU[1][tid] + sU[2][tid] + sU[3][tid];
        float tv = sT[tid];
        sPin[tid]           = u;
        sPin[D_EMB + tid]   = tv;
        sPin[2*D_EMB + tid] = u - tv;
        sPin[3*D_EMB + tid] = u * tv;
    }
    __syncthreads();

    // ---- prediction MLP: relu(pred_in @ W1p + b1p) @ W2p + b2p ----
    {
        float part = 0.0f;
        if (tid < H_A) {
            float hsum = b1p[tid];
            #pragma unroll 4
            for (int k = 0; k < K_IN; ++k)
                hsum += sPin[k] * W1p[k * H_A + tid];
            part = fmaxf(hsum, 0.0f) * W2p[tid];
        }
        sRed[tid] = part;
        __syncthreads();
        for (int off = 128; off > 0; off >>= 1) {
            if (tid < off) sRed[tid] += sRed[tid + off];
            __syncthreads();
        }
        if (tid == 0) out[b] = sRed[0] + b2p[0];
    }
}

// ---------------------------------------------------------------------------
extern "C" void kernel_launch(void* const* d_in, const int* in_sizes, int n_in,
                              void* d_out, int out_size, void* d_ws, size_t ws_size,
                              hipStream_t stream) {
    const int*   hist_ids  = (const int*)  d_in[0];
    const int*   hist_mask = (const int*)  d_in[1];
    const int*   tgt_ids   = (const int*)  d_in[2];
    const float* emb       = (const float*)d_in[3];
    const float* W1a       = (const float*)d_in[4];
    const float* b1a       = (const float*)d_in[5];
    const float* W2a       = (const float*)d_in[6];
    const float* b2a       = (const float*)d_in[7];
    const float* W1p       = (const float*)d_in[8];
    const float* b1p       = (const float*)d_in[9];
    const float* W2p       = (const float*)d_in[10];
    const float* b2p       = (const float*)d_in[11];
    float* out = (float*)d_out;

    _Float16* Wm    = (_Float16*)d_ws;                        // 128*128*2 = 32KB
    float*    Wdiff = (float*)((char*)d_ws + H_A * K_RED * 2); // 64*128*4  = 32KB

    const int B = in_sizes[2];                 // number of batch rows (4096)

    hipLaunchKernelGGL(din_prep_weights, dim3((H_A * K_RED + 255) / 256), dim3(256),
                       0, stream, W1a, Wm, Wdiff);
    hipLaunchKernelGGL(din_main, dim3(B), dim3(256), 0, stream,
                       hist_ids, hist_mask, tgt_ids, emb, b1a, W2a, b2a,
                       W1p, b1p, W2p, b2p, Wm, Wdiff, out);
}
